// LightningIndexer_8495445311707
// MI455X (gfx1250) — matrix operand order repaired and checked by reading, not compiled, not run on hardware
//
#include <hip/hip_runtime.h>

// ---------------------------------------------------------------------------
// LightningIndexer for MI455X (gfx1250, wave32, WMMA)
//   Prep   : fuse+transpose+pad weights -> WT bf16 [336][2048] (one-time, tiny)
//   Stage A: projection GEMM [B*T,2048] x [2048,336] -> Q(bf16), K(bf16), w(f32)
//            (7-way n-blocked; B loads batched ahead of the WMMA chain)
//   Stage B: per-head Q.K^T via v_wmma_f32_16x16x32_bf16, relu, head-weighted sum
// ---------------------------------------------------------------------------

typedef __bf16 bf16_t;
typedef bf16_t v16bf __attribute__((ext_vector_type(16)));
typedef bf16_t v8bf  __attribute__((ext_vector_type(8)));
typedef float  v8f   __attribute__((ext_vector_type(8)));
typedef float  v4f   __attribute__((ext_vector_type(4)));

#define WMMA_BF16 __builtin_amdgcn_wmma_f32_16x16x32_bf16

// Problem constants (match reference)
#define BB    2
#define TT    4096
#define DD    2048
#define HH    4
#define DI    64
#define NQ    (HH * DI)       // 256
#define NK    DI              // 64
#define NW    HH              // 4
#define NCOL  (NQ + NK + NW)  // 324
#define NPAD  336             // NCOL padded to 16
#define NTN   (NPAD / 16)     // 21 fused-N tiles
#define NB    7               // n-tiles per wave in stage A
#define NGRP  (NTN / NB)      // 3 n-groups

// ---------------------------------------------------------------------------
// Prep: WT[j][k] = fused transposed bf16 weights, zero-padded rows 324..335.
// ---------------------------------------------------------------------------
__global__ __launch_bounds__(256) void li_prep_kernel(
    const float* __restrict__ Wq, const float* __restrict__ Wk,
    const float* __restrict__ Ww, bf16_t* __restrict__ WT)
{
    const int k = blockIdx.x * 256 + threadIdx.x;   // 0..2047
    const int j = blockIdx.y;                       // 0..335
    float v;
    if      (j < NQ)      v = Wq[(size_t)k * NQ + j];
    else if (j < NQ + NK) v = Wk[(size_t)k * NK + (j - NQ)];
    else if (j < NCOL)    v = Ww[(size_t)k * NW + (j - NQ - NK)];
    else                  v = 0.0f;
    WT[(size_t)j * DD + k] = (bf16_t)v;
}

// ---------------------------------------------------------------------------
// Stage A: one wave = one 16-row m-tile x seven 16-col n-tiles.
// Per k-step: (1) issue ALL 7 B-fragment loads (kept live in Bf[]),
//             (2) A loads + f32->bf16 converts overlap them,
//             (3) 7 WMMAs drain with partial loadcnt waits.
// ---------------------------------------------------------------------------
__global__ __launch_bounds__(256) void li_proj_kernel(
    const float* __restrict__ x, const bf16_t* __restrict__ WT,
    bf16_t* __restrict__ Qo, bf16_t* __restrict__ Ko, float* __restrict__ Wto)
{
    const int lane   = threadIdx.x & 31;
    const int wave   = blockIdx.x * (blockDim.x >> 5) + (threadIdx.x >> 5);
    const int mtile  = wave / NGRP;         // 0..511  (B*T/16 row tiles)
    const int ng     = wave % NGRP;         // 0..2
    const int laneHi = lane >> 4;           // 0 or 1
    const int lane15 = lane & 15;

    const float* xrow = x + (size_t)(mtile * 16 + lane15) * DD;  // A row (M = lane15)

    int           jj[NB];                   // fused column per n-tile (N = lane15)
    const bf16_t* wtrow[NB];                // B column base, K-contiguous
    #pragma unroll
    for (int nt = 0; nt < NB; ++nt) {
        jj[nt]    = ng * (NB * 16) + nt * 16 + lane15;
        wtrow[nt] = WT + (size_t)jj[nt] * DD;
    }

    v8f acc[NB];
    #pragma unroll
    for (int nt = 0; nt < NB; ++nt) acc[nt] = (v8f){};

    for (int kb = 0; kb < DD; kb += 32) {
        // ---- (1) all 7 B fragments issued first, kept live
        const int kbb = kb + laneHi * 16;
        v16bf Bf[NB];
        #pragma unroll
        for (int nt = 0; nt < NB; ++nt)
            Bf[nt] = *(const v16bf*)(wtrow[nt] + kbb);

        // ---- (2) A fragment (16x32 bf16): lane-half K {0..15} / {16..31}
        const int ka = kb + laneHi * 16;
        v4f a0 = *(const v4f*)(xrow + ka);
        v4f a1 = *(const v4f*)(xrow + ka + 4);
        v4f a2 = *(const v4f*)(xrow + ka + 8);
        v4f a3 = *(const v4f*)(xrow + ka + 12);
        v16bf A;
        #pragma unroll
        for (int i = 0; i < 4; ++i) {
            A[i]      = (bf16_t)a0[i];
            A[4 + i]  = (bf16_t)a1[i];
            A[8 + i]  = (bf16_t)a2[i];
            A[12 + i] = (bf16_t)a3[i];
        }

        // ---- (3) 7 independent WMMA chains
        #pragma unroll
        for (int nt = 0; nt < NB; ++nt)
            acc[nt] = WMMA_BF16(false, A, false, Bf[nt], (short)0, acc[nt], false, false);
    }

    // C/D layout: VGPR r -> row (r + 8*laneHi), col lane15
    #pragma unroll
    for (int nt = 0; nt < NB; ++nt) {
        const int j = jj[nt];
        #pragma unroll
        for (int r = 0; r < 8; ++r) {
            const int   orow = mtile * 16 + laneHi * 8 + r;
            const float v    = acc[nt][r];
            if      (j < NQ)      Qo[(size_t)orow * NQ + j]              = (bf16_t)v;
            else if (j < NQ + NK) Ko[(size_t)orow * NK + (j - NQ)]       = (bf16_t)v;
            else if (j < NCOL)    Wto[(size_t)orow * NW + (j - NQ - NK)] = v;
        }
    }
}

// ---------------------------------------------------------------------------
// Stage B: one wave = one 16-row t-tile x one 256-col s-block.
// Q fragments (4 heads x 2 K-chunks) + per-row head weights held in VGPRs;
// inner loop over 16 s-tiles: load K tile, 8 WMMAs, relu + weighted head sum.
// Output streamed with non-temporal stores to keep K/Q resident in L2.
// ---------------------------------------------------------------------------
__global__ __launch_bounds__(256) void li_score_kernel(
    const bf16_t* __restrict__ Q, const bf16_t* __restrict__ K,
    const float* __restrict__ Wt, float* __restrict__ out)
{
    const int lane   = threadIdx.x & 31;
    const int wave   = blockIdx.x * (blockDim.x >> 5) + (threadIdx.x >> 5);
    const int ttile  = wave >> 4;          // 0..511
    const int sblk   = wave & 15;          // 0..15
    const int b      = ttile >> 8;
    const int t0     = (ttile & 255) * 16;
    const int laneHi = lane >> 4;
    const int lane15 = lane & 15;

    const bf16_t* Qb   = Q   + (size_t)b * TT * NQ;
    const bf16_t* Kb   = K   + (size_t)b * TT * NK;
    const float*  Wb   = Wt  + (size_t)b * TT * NW;
    float*        outb = out + (size_t)b * TT * TT;

    // ---- Q A-fragments: 4 heads x 2 DI-chunks, resident in VGPRs
    v16bf qa[HH][2];
    const bf16_t* qrow = Qb + (size_t)(t0 + lane15) * NQ;
    #pragma unroll
    for (int h = 0; h < HH; ++h) {
        #pragma unroll
        for (int c = 0; c < 2; ++c) {
            const int k0 = h * DI + c * 32 + laneHi * 16;
            v8bf lo = *(const v8bf*)(qrow + k0);
            v8bf hi = *(const v8bf*)(qrow + k0 + 8);
            v16bf f;
            #pragma unroll
            for (int i = 0; i < 8; ++i) { f[i] = lo[i]; f[8 + i] = hi[i]; }
            qa[h][c] = f;
        }
    }
    // ---- per-lane head weights for this lane's 8 output rows (b128 loads)
    v4f wrow[8];
    #pragma unroll
    for (int r = 0; r < 8; ++r)
        wrow[r] = *(const v4f*)(Wb + (size_t)(t0 + laneHi * 8 + r) * NW);

    const int s0base = sblk * 256;
    for (int st = 0; st < 16; ++st) {
        const int s0 = s0base + st * 16;
        const bf16_t* krow = Kb + (size_t)(s0 + lane15) * NK;
        // B fragments: lane = s-col, contiguous 16 K(di) values per lane-half
        v16bf kb0 = *(const v16bf*)(krow + laneHi * 16);
        v16bf kb1 = *(const v16bf*)(krow + 32 + laneHi * 16);
        // prefetch next s-tile's K rows (speculative, dropped if OOB)
        __builtin_prefetch(krow + 16 * NK, 0, 0);

        v8f acc = {};
        #pragma unroll
        for (int h = 0; h < HH; ++h) {
            v8f d = {};
            d = WMMA_BF16(false, qa[h][0], false, kb0, (short)0, d, false, false);
            d = WMMA_BF16(false, qa[h][1], false, kb1, (short)0, d, false, false);
            #pragma unroll
            for (int r = 0; r < 8; ++r) {
                float v = d[r];
                v = v > 0.0f ? v : 0.0f;                  // relu
                acc[r] = fmaf(wrow[r][h], v, acc[r]);     // head-weighted sum
            }
        }
        #pragma unroll
        for (int r = 0; r < 8; ++r)
            __builtin_nontemporal_store(
                acc[r], outb + (size_t)(t0 + laneHi * 8 + r) * TT + s0 + lane15);
    }
}

// ---------------------------------------------------------------------------
extern "C" void kernel_launch(void* const* d_in, const int* in_sizes, int n_in,
                              void* d_out, int out_size, void* d_ws, size_t ws_size,
                              hipStream_t stream) {
    const float* x  = (const float*)d_in[0];
    const float* Wq = (const float*)d_in[1];
    const float* Wk = (const float*)d_in[2];
    const float* Ww = (const float*)d_in[3];
    float* out = (float*)d_out;

    // Workspace layout (bytes):
    //   Q bf16 [B*T,256] | K bf16 [B*T,64] | w f32 [B*T,4] | WT bf16 [336][2048]
    char* ws = (char*)d_ws;
    size_t off = 0;
    bf16_t* Qo  = (bf16_t*)(ws + off); off += (size_t)BB * TT * NQ * 2;  // 4 MB
    bf16_t* Ko  = (bf16_t*)(ws + off); off += (size_t)BB * TT * NK * 2;  // 1 MB
    float*  Wto = (float*) (ws + off); off += (size_t)BB * TT * NW * 4;  // 128 KB
    bf16_t* WT  = (bf16_t*)(ws + off);                                   // 1.376 MB

    // Prep: fused transposed bf16 weights (336 x 2048)
    li_prep_kernel<<<dim3(DD / 256, NPAD), dim3(256), 0, stream>>>(Wq, Wk, Ww, WT);

    // Stage A: 512 m-tiles x 3 n-groups = 1536 waves = 192 blocks of 8 waves
    li_proj_kernel<<<dim3((BB * TT / 16) * NGRP / 8), dim3(256), 0, stream>>>(
        x, WT, Qo, Ko, Wto);

    // Stage B: 512 t-tiles x 16 s-blocks = 8192 waves = 1024 blocks of 8 waves
    li_score_kernel<<<dim3((BB * TT / 16) * 16 / 8), dim3(256), 0, stream>>>(
        Qo, Ko, Wto, out);
}
